// Transcoder_48936857370754
// MI455X (gfx1250) — matrix operand order repaired and checked by reading, not compiled, not run on hardware
//
#include <hip/hip_runtime.h>

// ---- plain trivially-constructible vector types (safe inside unions) ----
typedef __bf16        bf16x16 __attribute__((ext_vector_type(16)));
typedef float         f32x8   __attribute__((ext_vector_type(8)));
typedef float         f32x4   __attribute__((ext_vector_type(4)));
typedef unsigned int  u32x4   __attribute__((ext_vector_type(4)));

#define D_IN    768
#define N_ROWS  8192
#define N_FEAT  24576
#define D_OUT   768
#define TOPK    64
#define MROWS   32                  // rows per workgroup (2 x 16 row-tiles)
#define CHUNK   128                 // features per chunk = 8 waves * 16
#define NCHUNK  (N_FEAT / CHUNK)    // 192
#define KSTEPS  (D_IN / 32)         // 24 WMMA k-steps per tile
#define XS_STRIDE 776               // 768 + 8 ushort pad -> conflict-free ds_load_b128
#define ZPAD    132                 // zbuf row stride (floats), keeps 16B alignment
#define CANDCAP 256                 // per-row candidate list capacity
#define CTRIG   128                 // compact when count exceeds this

static __device__ __forceinline__ unsigned short f32_to_bf16(float f) {
    unsigned int u = __float_as_uint(f);
    u += 0x7fffu + ((u >> 16) & 1u);        // round to nearest even
    return (unsigned short)(u >> 16);
}

// ---------------------------------------------------------------------------
// Kernel 0: W_enc [768 x 24576] f32  ->  W_t [24576 x 768] bf16 (transposed)
// ---------------------------------------------------------------------------
__global__ void wenc_transpose_bf16(const float* __restrict__ W,
                                    unsigned short* __restrict__ Wt) {
    __shared__ float tile[32][33];
    const int fb = blockIdx.x * 32;
    const int kb = blockIdx.y * 32;
    const int tx = threadIdx.x & 31;
    const int ty = threadIdx.x >> 5;          // 0..7
#pragma unroll
    for (int i = 0; i < 4; ++i) {
        int k = ty + 8 * i;
        tile[k][tx] = W[(size_t)(kb + k) * N_FEAT + (fb + tx)];
    }
    __syncthreads();
#pragma unroll
    for (int i = 0; i < 4; ++i) {
        int f = ty + 8 * i;
        Wt[(size_t)(fb + f) * D_IN + (kb + tx)] = f32_to_bf16(tile[tx][f]);
    }
}

// ---------------------------------------------------------------------------
// Group helpers: 8-lane groups within a wave32 (rank k = lane & 7)
// ---------------------------------------------------------------------------
static __device__ __forceinline__ int group_excl_prefix(int v, int k) {
    int pre = v;
#pragma unroll
    for (int d = 1; d < 8; d <<= 1) {
        int t = __shfl_up(pre, d, 8);
        if (k >= d) pre += t;
    }
    return pre - v;                 // exclusive prefix within 8-lane group
}
static __device__ __forceinline__ int group_sum(int v) {
#pragma unroll
    for (int d = 1; d < 8; d <<= 1) v += __shfl_xor(v, d, 8);
    return v;
}

// Binary search on positive-float bit patterns for a lower bound of the
// TOPK-th largest among this group's candidate registers (invalid slots = 0).
static __device__ __forceinline__ unsigned kth_lower_bound(const unsigned* vb,
                                                           int nIter) {
    unsigned lo = 0u, hi = 0x7F800000u;
    for (int it = 0; it < nIter; ++it) {
        unsigned p = (lo + hi) >> 1;
        int cc = 0;
#pragma unroll
        for (int j = 0; j < 32; ++j) cc += (vb[j] >= p) ? 1 : 0;
        cc = group_sum(cc);
        if (cc >= TOPK) lo = p; else hi = p;
    }
    return lo ? lo : 1u;            // never keep the zero (invalid) slots
}

// ---------------------------------------------------------------------------
// Kernel 1: fused encoder-GEMM (bf16 WMMA) + batched lazy-threshold top-64 +
//           sparse decoder.  One workgroup = 32 rows, 8 wave32s.
// ---------------------------------------------------------------------------
__global__ void __launch_bounds__(256)
sae_encode_topk_decode(const unsigned short* __restrict__ Wt,   // [24576][768] bf16
                       const float* __restrict__ x,             // [8192][768]  f32
                       const float* __restrict__ b_enc,         // [24576]
                       const float* __restrict__ W_dec,         // [24576][768]
                       const float* __restrict__ b_dec,         // [768]
                       float* __restrict__ out)                 // [8192][768]
{
    __shared__ unsigned short xs[MROWS * XS_STRIDE];   // 48.5 KB bf16 x-tile
    __shared__ float zbuf[2][MROWS][ZPAD];             // 33 KB double-buffered z
    __shared__ float          candv[MROWS][CANDCAP];   // 32 KB candidate values
    __shared__ unsigned short candi[MROWS][CANDCAP];   // 16 KB candidate indices
    __shared__ float topv[MROWS][TOPK];                // 8 KB
    __shared__ int   topi[MROWS][TOPK];                // 8 KB
    __shared__ int   cnt[MROWS];
    __shared__ float th[MROWS];                        // lazy lower-bound threshold

    const int tid     = threadIdx.x;
    const int wave    = tid >> 5;
    const int lane    = tid & 31;
    const int n       = lane & 15;   // tile col (B/C) / tile row (A)
    const int half    = lane >> 4;   // lane-group select per ISA WMMA layouts
    const int grp     = lane >> 3;   // selection: which of this wave's 4 rows
    const int k8      = lane & 7;    // selection: rank within 8-lane group
    const int selrow  = 4 * wave + grp;
    const int rowbase = blockIdx.x * MROWS;

    // ---- stage x tile: f32 -> bf16 into padded LDS rows ----
    for (int i = tid; i < MROWS * D_IN; i += 256) {
        int r = i / D_IN, c = i - r * D_IN;
        xs[r * XS_STRIDE + c] = f32_to_bf16(x[(size_t)(rowbase + r) * D_IN + c]);
    }
    for (int i = tid; i < MROWS * TOPK; i += 256) {
        (&topv[0][0])[i] = 0.0f;
        (&topi[0][0])[i] = 0;
    }
    if (tid < MROWS) { cnt[tid] = 0; th[tid] = 0.0f; }
    __syncthreads();

    const unsigned short* a0base = xs + n * XS_STRIDE + half * 8;
    const unsigned short* a1base = xs + (16 + n) * XS_STRIDE + half * 8;

    for (int ch = 0; ch < NCHUNK; ++ch) {
        // ==== compute: two 16x16 z tiles, one B fragment feeds both ====
        const int fbase = ch * CHUNK + wave * 16;
        f32x8 acc0 = {0.f, 0.f, 0.f, 0.f, 0.f, 0.f, 0.f, 0.f};
        f32x8 acc1 = acc0;
        const unsigned short* bcol = Wt + (size_t)(fbase + n) * D_IN + half * 8;
#pragma unroll 4
        for (int ks = 0; ks < KSTEPS; ++ks) {
            union { u32x4 q[2]; bf16x16 v; } A0, A1, B;
            B.q[0]  = *(const u32x4*)(bcol + ks * 32);
            B.q[1]  = *(const u32x4*)(bcol + ks * 32 + 16);
            A0.q[0] = *(const u32x4*)(a0base + ks * 32);
            A0.q[1] = *(const u32x4*)(a0base + ks * 32 + 16);
            A1.q[0] = *(const u32x4*)(a1base + ks * 32);
            A1.q[1] = *(const u32x4*)(a1base + ks * 32 + 16);
            acc0 = __builtin_amdgcn_wmma_f32_16x16x32_bf16(
                false, A0.v, false, B.v, (short)0, acc0, false, false);
            acc1 = __builtin_amdgcn_wmma_f32_16x16x32_bf16(
                false, A1.v, false, B.v, (short)0, acc1, false, false);
        }
        const float be  = b_enc[fbase + n];
        const int   buf = ch & 1;
#pragma unroll
        for (int r = 0; r < 8; ++r) {
            zbuf[buf][r + 8 * half][wave * 16 + n]      = acc0[r] + be;
            zbuf[buf][16 + r + 8 * half][wave * 16 + n] = acc1[r] + be;
        }
        __syncthreads();

        // ==== select: every wave filters its own 4 rows (8 lanes/row) ====
        // (reads zbuf[buf]; overlaps with next iteration's compute into !buf)
        {
            const float* zr = &zbuf[buf][selrow][16 * k8];
            float t  = th[selrow];
            int   c0 = cnt[selrow];
            float v[16];
            *(f32x4*)&v[0]  = *(const f32x4*)(zr + 0);
            *(f32x4*)&v[4]  = *(const f32x4*)(zr + 4);
            *(f32x4*)&v[8]  = *(const f32x4*)(zr + 8);
            *(f32x4*)&v[12] = *(const f32x4*)(zr + 12);
            unsigned m = 0;
#pragma unroll
            for (int e = 0; e < 16; ++e) m |= (v[e] > t) ? (1u << e) : 0u;
            const int lc   = __popc(m);
            const int pre  = group_excl_prefix(lc, k8);
            const int tot  = group_sum(lc);
            const int base = c0 + pre;
            if (m) {
#pragma unroll
                for (int e = 0; e < 16; ++e) {
                    if ((m >> e) & 1u) {
                        int p = base + __popc(m & ((1u << e) - 1u));
                        if (p < CANDCAP) {
                            candv[selrow][p] = v[e];
                            candi[selrow][p] =
                                (unsigned short)(ch * CHUNK + 16 * k8 + e);
                        }
                    }
                }
            }
            int nc = c0 + tot; if (nc > CANDCAP) nc = CANDCAP;
            if (k8 == 0) cnt[selrow] = nc;

            // ==== compact when the list is more than half full ====
            if (nc > CTRIG) {
                const int basej = k8 * 32;
                unsigned vb[32];
#pragma unroll
                for (int j = 0; j < 32; j += 4) {
                    u32x4 q = *(const u32x4*)(&candv[selrow][basej + j]);
                    vb[j] = q[0]; vb[j+1] = q[1]; vb[j+2] = q[2]; vb[j+3] = q[3];
                }
                unsigned iw[16];
#pragma unroll
                for (int j = 0; j < 16; j += 4) {
                    u32x4 q = *(const u32x4*)(((const unsigned*)&candi[selrow][basej]) + j);
                    iw[j] = q[0]; iw[j+1] = q[1]; iw[j+2] = q[2]; iw[j+3] = q[3];
                }
#pragma unroll
                for (int j = 0; j < 32; ++j)
                    if (basej + j >= nc) vb[j] = 0u;

                const unsigned kth = kth_lower_bound(vb, 16);

                int lk = 0;
#pragma unroll
                for (int j = 0; j < 32; ++j) lk += (vb[j] >= kth) ? 1 : 0;
                int pos = group_excl_prefix(lk, k8);
                int tk  = group_sum(lk);
#pragma unroll
                for (int j = 0; j < 32; ++j) {
                    if (vb[j] >= kth) {
                        if (pos < CANDCAP) {
                            candv[selrow][pos] = __uint_as_float(vb[j]);
                            candi[selrow][pos] = (unsigned short)
                                ((iw[j >> 1] >> ((j & 1) * 16)) & 0xFFFFu);
                        }
                        ++pos;
                    }
                }
                if (k8 == 0) {
                    cnt[selrow] = tk < CANDCAP ? tk : CANDCAP;
                    th[selrow]  = __uint_as_float(kth);
                }
            }
        }
        // no extra barrier: next iteration's compute uses the other zbuf,
        // and its end-of-compute barrier orders everything before select(ch+1)
    }
    __syncthreads();

    // ==== final: exact-ish top-64 per row from its candidate list ====
    {
        const int c     = cnt[selrow];
        const int basej = k8 * 32;
        unsigned vb[32];
#pragma unroll
        for (int j = 0; j < 32; j += 4) {
            u32x4 q = *(const u32x4*)(&candv[selrow][basej + j]);
            vb[j] = q[0]; vb[j+1] = q[1]; vb[j+2] = q[2]; vb[j+3] = q[3];
        }
        unsigned iw[16];
#pragma unroll
        for (int j = 0; j < 16; j += 4) {
            u32x4 q = *(const u32x4*)(((const unsigned*)&candi[selrow][basej]) + j);
            iw[j] = q[0]; iw[j+1] = q[1]; iw[j+2] = q[2]; iw[j+3] = q[3];
        }
#pragma unroll
        for (int j = 0; j < 32; ++j)
            if (basej + j >= c) vb[j] = 0u;

        const unsigned kth = kth_lower_bound(vb, 26);

        int lk = 0;
#pragma unroll
        for (int j = 0; j < 32; ++j) lk += (vb[j] >= kth) ? 1 : 0;
        int pos = group_excl_prefix(lk, k8);
#pragma unroll
        for (int j = 0; j < 32; ++j) {
            if (vb[j] >= kth) {
                if (pos < TOPK) {
                    topv[selrow][pos] = __uint_as_float(vb[j]);
                    topi[selrow][pos] =
                        (int)((iw[j >> 1] >> ((j & 1) * 16)) & 0xFFFFu);
                }
                ++pos;
            }
        }
    }
    __syncthreads();

    // ==== fused sparse decoder: out = relu(topv) @ W_dec[topi] + b_dec ====
    const float bd0 = b_dec[tid];
    const float bd1 = b_dec[tid + 256];
    const float bd2 = b_dec[tid + 512];
    for (int r = 0; r < MROWS; ++r) {
        float a0 = bd0, a1 = bd1, a2 = bd2;
        for (int s = 0; s < TOPK; ++s) {
            float v = topv[r][s];                   // uniform LDS broadcast
            if (v > 0.0f) {
                const float* wd = W_dec + (size_t)topi[r][s] * D_OUT;
                a0 = fmaf(v, wd[tid],       a0);
                a1 = fmaf(v, wd[tid + 256], a1);
                a2 = fmaf(v, wd[tid + 512], a2);
            }
        }
        float* o = out + (size_t)(rowbase + r) * D_OUT;
        o[tid] = a0; o[tid + 256] = a1; o[tid + 512] = a2;
    }
}

// ---------------------------------------------------------------------------
extern "C" void kernel_launch(void* const* d_in, const int* in_sizes, int n_in,
                              void* d_out, int out_size, void* d_ws, size_t ws_size,
                              hipStream_t stream) {
    const float* x     = (const float*)d_in[0];
    const float* W_enc = (const float*)d_in[1];
    const float* b_enc = (const float*)d_in[2];
    const float* W_dec = (const float*)d_in[3];
    const float* b_dec = (const float*)d_in[4];
    float* out = (float*)d_out;

    // workspace: transposed bf16 encoder weights, 24576*768*2 = 37,748,736 B
    unsigned short* Wt = (unsigned short*)d_ws;

    dim3 g0(N_FEAT / 32, D_IN / 32);                 // 768 x 24 tiles
    wenc_transpose_bf16<<<g0, 256, 0, stream>>>(W_enc, Wt);

    sae_encode_topk_decode<<<N_ROWS / MROWS, 256, 0, stream>>>(
        Wt, x, b_enc, W_dec, b_dec, out);
}
